// Attention_20529943675070
// MI455X (gfx1250) — matrix operand order repaired and checked by reading, not compiled, not run on hardware
//
#include <hip/hip_runtime.h>

// ---------------------------------------------------------------------------
// MHA forward for MI455X (gfx1250, wave32, WMMA bf16 16x16x32, fp32 accum).
//   x  [2,2048,1024] f32
//   Wq [1024,512] f32, Wkv [1024,1024] f32, Wo [512,1024] f32, bo [1024] f32
//   out [2,2048,1024] f32
// Workspace: Q bf16 [b,h,n,d] (pre-scaled by 1/8), K bf16 [b,h,n,d],
//            Vt bf16 [b,h,d,n], Ab bf16 [b*n, h*d].
// ---------------------------------------------------------------------------

typedef __attribute__((ext_vector_type(16))) __bf16 v16bf;
typedef __attribute__((ext_vector_type(8)))  __bf16 v8bf;
typedef __attribute__((ext_vector_type(8)))  float  v8f;

union BF16x16 { v16bf v; v8bf h[2]; };

__device__ __forceinline__ v8f wmma_bf16(v16bf a, v16bf b, v8f c) {
  // D(f32 16x16) = A(bf16 16x32) * B(bf16 32x16) + C
  return __builtin_amdgcn_wmma_f32_16x16x32_bf16(
      /*neg_a=*/false, a, /*neg_b=*/false, b,
      /*c_mod=*/(short)0, c, /*reuse_a=*/false, /*reuse_b=*/false);
}

__device__ __forceinline__ v8bf cvt8(float4 a, float4 b) {
  v8bf r;
  r[0] = (__bf16)a.x; r[1] = (__bf16)a.y; r[2] = (__bf16)a.z; r[3] = (__bf16)a.w;
  r[4] = (__bf16)b.x; r[5] = (__bf16)b.y; r[6] = (__bf16)b.z; r[7] = (__bf16)b.w;
  return r;
}

// ---------------------------------------------------------------------------
// Kernel 1: fused QKV projection. Grid (64 Mtiles, 24 Ntiles). 128 threads.
// Block tile: 64(M) x 64(N), K-loop of 32. Wave w owns rows [w*16, w*16+16).
// ---------------------------------------------------------------------------
__global__ __launch_bounds__(128) void qkv_proj_kernel(
    const float* __restrict__ x, const float* __restrict__ Wq,
    const float* __restrict__ Wkv,
    __bf16* __restrict__ Qb, __bf16* __restrict__ Kb, __bf16* __restrict__ Vt)
{
  __shared__ __bf16 Bs[64 * 40];  // B tile transposed: Bs[n][k], pitch 40 bf16

  const int tid  = threadIdx.x;
  const int lane = tid & 31, wave = tid >> 5;
  const int g = lane >> 4, ln = lane & 15;
  const int mbase = blockIdx.x * 64;
  const int yb    = blockIdx.y;  // 0..7 -> Q, 8..15 -> K, 16..23 -> V

  const float* Wp; int ld, ncol0, region;
  if (yb < 8)       { Wp = Wq;  ld = 512;  ncol0 = yb * 64;       region = 0; }
  else if (yb < 16) { Wp = Wkv; ld = 1024; ncol0 = (yb - 8) * 64; region = 1; }
  else              { Wp = Wkv; ld = 1024; ncol0 = (yb - 8) * 64; region = 2; }

  v8f acc[4];
  #pragma unroll
  for (int nt = 0; nt < 4; ++nt)
    #pragma unroll
    for (int r = 0; r < 8; ++r) acc[nt][r] = 0.0f;

  const int arow = mbase + wave * 16 + ln;

  for (int kb = 0; kb < 1024; kb += 32) {
    __syncthreads();
    // Cooperative load of W[kb:kb+32, ncol0:ncol0+64] -> transposed bf16 LDS
    #pragma unroll
    for (int p = 0; p < 4; ++p) {
      int idx = p * 128 + tid;        // 0..511
      int kk  = idx >> 4;             // 0..31
      int nn  = (idx & 15) * 4;       // 0..60
      float4 w = *(const float4*)(Wp + (size_t)(kb + kk) * ld + ncol0 + nn);
      Bs[(nn + 0) * 40 + kk] = (__bf16)w.x;
      Bs[(nn + 1) * 40 + kk] = (__bf16)w.y;
      Bs[(nn + 2) * 40 + kk] = (__bf16)w.z;
      Bs[(nn + 3) * 40 + kk] = (__bf16)w.w;
    }
    // Prefetch next weight tile (global_prefetch_b8) while WMMAs run.
    if (kb + 32 < 1024) {
      int kk = tid >> 4, nn = (tid & 15) * 4;
      __builtin_prefetch(Wp + (size_t)(kb + 32 + kk) * ld + ncol0 + nn, 0, 3);
    }
    __syncthreads();

    // A fragment straight from global fp32 (row ln, K halves 8g..8g+7 / +16)
    const float* ap = x + (size_t)arow * 1024 + kb;
    float4 f0 = *(const float4*)(ap + g * 8);
    float4 f1 = *(const float4*)(ap + g * 8 + 4);
    float4 f2 = *(const float4*)(ap + 16 + g * 8);
    float4 f3 = *(const float4*)(ap + 16 + g * 8 + 4);
    BF16x16 ua; ua.h[0] = cvt8(f0, f1); ua.h[1] = cvt8(f2, f3);

    #pragma unroll
    for (int nt = 0; nt < 4; ++nt) {
      int nc = nt * 16 + ln;
      BF16x16 ub;
      ub.h[0] = *(const v8bf*)&Bs[nc * 40 + g * 8];
      ub.h[1] = *(const v8bf*)&Bs[nc * 40 + 16 + g * 8];
      acc[nt] = wmma_bf16(ua.v, ub.v, acc[nt]);
    }
  }

  // Scatter to Q (pre-scaled), K [n,d] and V transposed [d,n], all bf16.
  // Region test is block-uniform: keep ONE branch, not per-element branches.
  const int rowbase = mbase + wave * 16;
  const int b  = rowbase >> 11;         // batch (tiles never straddle)
  const int n0 = rowbase & 2047;
  const int h  = yb & 7;                // head id within region
  const size_t ho = (size_t)(b * 8 + h) * (size_t)(2048 * 64);

  if (region == 2) {
    // V transposed: lane's 8 rows are contiguous along n -> one 16B store.
    #pragma unroll
    for (int nt = 0; nt < 4; ++nt) {
      int dd = nt * 16 + ln;
      v8bf pk;
      #pragma unroll
      for (int r = 0; r < 8; ++r) pk[r] = (__bf16)acc[nt][r];
      *(v8bf*)(Vt + ho + (size_t)dd * 2048 + n0 + 8 * g) = pk;
    }
  } else {
    __bf16* dst = (region == 0) ? Qb : Kb;
    const float sc = (region == 0) ? 0.125f : 1.0f;  // 1/sqrt(64), exact
    #pragma unroll
    for (int r = 0; r < 8; ++r) {
      __bf16* rowp = dst + ho + (size_t)(n0 + r + 8 * g) * 64 + ln;
      #pragma unroll
      for (int nt = 0; nt < 4; ++nt)
        rowp[nt * 16] = (__bf16)(acc[nt][r] * sc);
    }
  }
}

// ---------------------------------------------------------------------------
// Kernel 2: flash attention. Grid (32 qtiles, 8 heads, 2 batch). 128 threads.
// Each wave: 16 query rows, online softmax over 64-key tiles, WMMA for S & PV.
// ---------------------------------------------------------------------------
__global__ __launch_bounds__(128) void flash_attn_kernel(
    const __bf16* __restrict__ Qb, const __bf16* __restrict__ Kb,
    const __bf16* __restrict__ Vt, __bf16* __restrict__ Ab)
{
  __shared__ __bf16 Plds[4 * 16 * 72];  // per-wave 16x64 P tile, pitch 72

  const int tid  = threadIdx.x;
  const int lane = tid & 31, wave = tid >> 5;
  const int g = lane >> 4, ln = lane & 15;
  const int h = blockIdx.y, b = blockIdx.z;
  const size_t ho = (size_t)(b * 8 + h) * (size_t)(2048 * 64);
  const __bf16* Qh = Qb + ho;
  const __bf16* Kh = Kb + ho;
  const __bf16* Vh = Vt + ho;   // [64][2048]

  const int q0 = blockIdx.x * 64 + wave * 16;

  // Q A-fragments (d chunks 0..31, 32..63) kept in registers all loop long.
  v16bf qf[2];
  #pragma unroll
  for (int c = 0; c < 2; ++c) {
    const __bf16* qp = Qh + (size_t)(q0 + ln) * 64 + c * 32 + g * 16;
    BF16x16 u;
    u.h[0] = *(const v8bf*)qp;
    u.h[1] = *(const v8bf*)(qp + 8);
    qf[c] = u.v;
  }

  float m_i[8], l_i[8];
  v8f o[4];
  #pragma unroll
  for (int r = 0; r < 8; ++r) { m_i[r] = -1e30f; l_i[r] = 0.0f; }
  #pragma unroll
  for (int dt = 0; dt < 4; ++dt)
    #pragma unroll
    for (int r = 0; r < 8; ++r) o[dt][r] = 0.0f;

  const int pbase = wave * 16 * 72;

  for (int kb = 0; kb < 2048; kb += 64) {
    // Prefetch next key tile rows while this tile computes.
    if (kb + 64 < 2048)
      __builtin_prefetch(Kh + (size_t)(kb + 64 + lane * 2) * 64, 0, 3);

    // ---- S = Q K^T (scale pre-folded into Q) : 4 n-tiles x 2 d-chunks
    v8f s[4];
    #pragma unroll
    for (int nt = 0; nt < 4; ++nt) {
      v8f a;
      #pragma unroll
      for (int r = 0; r < 8; ++r) a[r] = 0.0f;
      #pragma unroll
      for (int c = 0; c < 2; ++c) {
        const v16bf* kp = (const v16bf*)(Kh +
            (size_t)(kb + nt * 16 + ln) * 64 + c * 32 + g * 16);
        a = wmma_bf16(qf[c], *kp, a);
      }
      s[nt] = a;
    }

    // ---- online softmax: row max over 4 tiles + 16-lane butterfly
    float rm[8];
    #pragma unroll
    for (int r = 0; r < 8; ++r)
      rm[r] = fmaxf(fmaxf(s[0][r], s[1][r]), fmaxf(s[2][r], s[3][r]));
    #pragma unroll
    for (int mask = 1; mask < 16; mask <<= 1)
      #pragma unroll
      for (int r = 0; r < 8; ++r)
        rm[r] = fmaxf(rm[r], __shfl_xor(rm[r], mask, 16));

    float alpha[8], rs[8];
    #pragma unroll
    for (int r = 0; r < 8; ++r) {
      float mn = fmaxf(m_i[r], rm[r]);
      alpha[r] = __expf(m_i[r] - mn);
      m_i[r] = mn;
      rs[r] = 0.0f;
    }

    // ---- P = exp(S - m), stash as bf16 in wave-private LDS (A-layout source)
    #pragma unroll
    for (int nt = 0; nt < 4; ++nt)
      #pragma unroll
      for (int r = 0; r < 8; ++r) {
        float p = __expf(s[nt][r] - m_i[r]);
        rs[r] += p;
        Plds[pbase + (r + 8 * g) * 72 + nt * 16 + ln] = (__bf16)p;
      }

    #pragma unroll
    for (int mask = 1; mask < 16; mask <<= 1)
      #pragma unroll
      for (int r = 0; r < 8; ++r)
        rs[r] += __shfl_xor(rs[r], mask, 16);

    #pragma unroll
    for (int r = 0; r < 8; ++r) l_i[r] = l_i[r] * alpha[r] + rs[r];

    #pragma unroll
    for (int dt = 0; dt < 4; ++dt)
      #pragma unroll
      for (int r = 0; r < 8; ++r) o[dt][r] *= alpha[r];

    __syncthreads();

    // ---- O += P V : P A-fragments from LDS, V B-fragments from Vt (contig)
    v16bf pf[2];
    #pragma unroll
    for (int c = 0; c < 2; ++c) {
      const __bf16* pp = &Plds[pbase + ln * 72 + c * 32 + g * 16];
      BF16x16 u;
      u.h[0] = *(const v8bf*)pp;
      u.h[1] = *(const v8bf*)(pp + 8);
      pf[c] = u.v;
    }
    #pragma unroll
    for (int dt = 0; dt < 4; ++dt) {
      #pragma unroll
      for (int c = 0; c < 2; ++c) {
        const v16bf* vp = (const v16bf*)(Vh +
            (size_t)(dt * 16 + ln) * 2048 + kb + c * 32 + g * 16);
        o[dt] = wmma_bf16(pf[c], *vp, o[dt]);
      }
    }
    __syncthreads();
  }

  // ---- finalize and store merged-head bf16 activations [b*n, h*64+d]
  float inv[8];
  #pragma unroll
  for (int r = 0; r < 8; ++r) inv[r] = 1.0f / l_i[r];

  #pragma unroll
  for (int dt = 0; dt < 4; ++dt) {
    #pragma unroll
    for (int r = 0; r < 8; ++r) {
      int row = b * 2048 + q0 + r + 8 * g;
      int col = h * 64 + dt * 16 + ln;
      Ab[(size_t)row * 512 + col] = (__bf16)(o[dt][r] * inv[r]);
    }
  }
}

// ---------------------------------------------------------------------------
// Kernel 3: output projection + bias. Grid (64 Mtiles, 16 Ntiles). 128 thrds.
// ---------------------------------------------------------------------------
__global__ __launch_bounds__(128) void out_proj_kernel(
    const __bf16* __restrict__ A, const float* __restrict__ Wo,
    const float* __restrict__ bo, float* __restrict__ out)
{
  __shared__ __bf16 Bs[64 * 40];

  const int tid  = threadIdx.x;
  const int lane = tid & 31, wave = tid >> 5;
  const int g = lane >> 4, ln = lane & 15;
  const int mbase = blockIdx.x * 64;
  const int ncol0 = blockIdx.y * 64;

  v8f acc[4];
  #pragma unroll
  for (int nt = 0; nt < 4; ++nt)
    #pragma unroll
    for (int r = 0; r < 8; ++r) acc[nt][r] = 0.0f;

  const int arow = mbase + wave * 16 + ln;

  for (int kb = 0; kb < 512; kb += 32) {
    __syncthreads();
    #pragma unroll
    for (int p = 0; p < 4; ++p) {
      int idx = p * 128 + tid;
      int kk  = idx >> 4;
      int nn  = (idx & 15) * 4;
      float4 w = *(const float4*)(Wo + (size_t)(kb + kk) * 1024 + ncol0 + nn);
      Bs[(nn + 0) * 40 + kk] = (__bf16)w.x;
      Bs[(nn + 1) * 40 + kk] = (__bf16)w.y;
      Bs[(nn + 2) * 40 + kk] = (__bf16)w.z;
      Bs[(nn + 3) * 40 + kk] = (__bf16)w.w;
    }
    if (kb + 32 < 512) {
      int kk = tid >> 4, nn = (tid & 15) * 4;
      __builtin_prefetch(Wo + (size_t)(kb + 32 + kk) * 1024 + ncol0 + nn, 0, 3);
    }
    __syncthreads();

    const __bf16* ap = A + (size_t)arow * 512 + kb;
    BF16x16 ua;
    ua.h[0] = *(const v8bf*)(ap + g * 8);
    ua.h[1] = *(const v8bf*)(ap + 16 + g * 8);

    #pragma unroll
    for (int nt = 0; nt < 4; ++nt) {
      int nc = nt * 16 + ln;
      BF16x16 ub;
      ub.h[0] = *(const v8bf*)&Bs[nc * 40 + g * 8];
      ub.h[1] = *(const v8bf*)&Bs[nc * 40 + 16 + g * 8];
      acc[nt] = wmma_bf16(ua.v, ub.v, acc[nt]);
    }
  }

  #pragma unroll
  for (int nt = 0; nt < 4; ++nt) {
    float bias = bo[ncol0 + nt * 16 + ln];
    #pragma unroll
    for (int r = 0; r < 8; ++r) {
      size_t row = (size_t)(mbase + wave * 16 + r + 8 * g);
      out[row * 1024 + ncol0 + nt * 16 + ln] = acc[nt][r] + bias;
    }
  }
}

// ---------------------------------------------------------------------------
extern "C" void kernel_launch(void* const* d_in, const int* in_sizes, int n_in,
                              void* d_out, int out_size, void* d_ws, size_t ws_size,
                              hipStream_t stream) {
  (void)in_sizes; (void)n_in; (void)out_size; (void)ws_size;

  const float* x   = (const float*)d_in[0];
  const float* Wq  = (const float*)d_in[1];
  const float* Wkv = (const float*)d_in[2];
  const float* Wo  = (const float*)d_in[3];
  const float* bo  = (const float*)d_in[4];
  float* out = (float*)d_out;

  char* ws = (char*)d_ws;
  const size_t SEG = (size_t)2 * 8 * 2048 * 64 * 2;  // 4 MB per bf16 tensor
  __bf16* Qb = (__bf16*)(ws);
  __bf16* Kb = (__bf16*)(ws + SEG);
  __bf16* Vt = (__bf16*)(ws + 2 * SEG);
  __bf16* Ab = (__bf16*)(ws + 3 * SEG);

  qkv_proj_kernel<<<dim3(64, 24, 1), 128, 0, stream>>>(x, Wq, Wkv, Qb, Kb, Vt);
  flash_attn_kernel<<<dim3(32, 8, 2), 128, 0, stream>>>(Qb, Kb, Vt, Ab);
  out_proj_kernel<<<dim3(64, 16, 1), 128, 0, stream>>>(Ab, Wo, bo, out);
}
